// ConditionNetDDFT_49813030699480
// MI455X (gfx1250) — matrix operand order repaired and checked by reading, not compiled, not run on hardware
//
#include <hip/hip_runtime.h>

#define HW    65536   // 256*256
#define BATCH 16
#define PSTRIDE 520   // per-sample folded params: K1(192) c1(64) c2(64) K3(192) c3(3) pad

typedef __attribute__((ext_vector_type(16))) _Float16 v16h;
typedef __attribute__((ext_vector_type(8)))  float    v8f;

__device__ __forceinline__ float lrelu_f(float x) { return x >= 0.f ? x : 0.2f * x; }

// ---------------------------------------------------------------------------
// avgpool(3,2,1, count_include_pad) of per-(b,c)-affine-transformed input.
// pooled = (a*sum_valid + t*cnt) / 9   (pad zeros contribute nothing)
// ---------------------------------------------------------------------------
__global__ void pool_affine_kernel(const float* __restrict__ X, float* __restrict__ P,
                                   int C, int Hin, int Win, int Ho, int Wo,
                                   const float* __restrict__ aff_a,
                                   const float* __restrict__ aff_t)
{
    int total = BATCH * C * Ho * Wo;
    int idx = blockIdx.x * blockDim.x + threadIdx.x;
    if (idx >= total) return;
    int ox = idx % Wo; int t1 = idx / Wo;
    int oy = t1 % Ho;  int t2 = t1 / Ho;
    int c  = t2 % C;   int b  = t2 / C;
    int y0 = 2 * oy - 1, x0 = 2 * ox - 1;
    const float* xp = X + (size_t)(b * C + c) * Hin * Win;
    float s = 0.f; int cnt = 0;
    #pragma unroll
    for (int dy = 0; dy < 3; ++dy) {
        int y = y0 + dy;
        if ((unsigned)y >= (unsigned)Hin) continue;
        #pragma unroll
        for (int dx = 0; dx < 3; ++dx) {
            int x = x0 + dx;
            if ((unsigned)x >= (unsigned)Win) continue;
            s += xp[y * Win + x]; cnt++;
        }
    }
    float a = 1.f, t = 0.f;
    if (aff_a) { a = aff_a[b * C + c]; t = aff_t[b * C + c]; }
    P[idx] = (a * s + t * (float)cnt) * (1.f / 9.f);
}

// ---------------------------------------------------------------------------
// Y = lrelu(W @ P + bias*cnt/9)  (bias was inside the pre-pool conv in the ref,
// so it gets the pooling valid-count weight). Optionally accumulate per-(b,co)
// sum / sumsq for the following instance norm.
// ---------------------------------------------------------------------------
__global__ void conv_lrelu_stats_kernel(const float* __restrict__ P,
                                        const float* __restrict__ W,
                                        const float* __restrict__ bias,
                                        float* __restrict__ Y,
                                        int Ci, int Co, int Ho, int Wo, int Hin, int Win,
                                        float* __restrict__ sums, float* __restrict__ ssqs)
{
    int co = blockIdx.y, b = blockIdx.z;
    int npix = Ho * Wo;
    int pix = blockIdx.x * blockDim.x + threadIdx.x;
    float val = 0.f;
    if (pix < npix) {
        int oy = pix / Wo, ox = pix % Wo;
        int ylo = 2 * oy - 1; if (ylo < 0) ylo = 0;
        int yhi = 2 * oy + 1; if (yhi > Hin - 1) yhi = Hin - 1;
        int xlo = 2 * ox - 1; if (xlo < 0) xlo = 0;
        int xhi = 2 * ox + 1; if (xhi > Win - 1) xhi = Win - 1;
        float cnt = (float)((yhi - ylo + 1) * (xhi - xlo + 1));
        float acc = bias[co] * (cnt * (1.f / 9.f));
        const float* wr = W + co * Ci;
        const float* pp = P + (size_t)b * Ci * npix + pix;
        for (int ci = 0; ci < Ci; ++ci)
            acc = fmaf(wr[ci], pp[(size_t)ci * npix], acc);
        val = lrelu_f(acc);
        Y[(size_t)(b * Co + co) * npix + pix] = val;
    }
    if (sums) {
        __shared__ float r1[256], r2[256];
        int tid = threadIdx.x;
        r1[tid] = val; r2[tid] = val * val;
        __syncthreads();
        for (int s = 128; s > 0; s >>= 1) {
            if (tid < s) { r1[tid] += r1[tid + s]; r2[tid] += r2[tid + s]; }
            __syncthreads();
        }
        if (tid == 0) {
            atomicAdd(&sums[b * Co + co], r1[0]);
            atomicAdd(&ssqs[b * Co + co], r2[0]);
        }
    }
}

// InstanceNorm -> per-(b,c) affine (a,t) folded into next stage's pool input.
__global__ void inorm_prep_kernel(const float* __restrict__ sums, const float* __restrict__ ssqs,
                                  const float* __restrict__ g, const float* __restrict__ e,
                                  float* __restrict__ aff_a, float* __restrict__ aff_t,
                                  int C, int npix)
{
    int idx = blockIdx.x * blockDim.x + threadIdx.x;
    if (idx >= BATCH * C) return;
    int c = idx % C;
    float inv_n = 1.f / (float)npix;
    float m = sums[idx] * inv_n;
    float v = ssqs[idx] * inv_n - m * m;
    float a = g[c] / sqrtf(v + 1e-5f);
    aff_a[idx] = a;
    aff_t[idx] = e[c] - m * a;
}

// fea[b] = wf @ mean_pix(Y4[b]) + bf     (Y4: [B,128,8,8])
__global__ void fea_kernel(const float* __restrict__ Y4, const float* __restrict__ wf,
                           const float* __restrict__ bf, float* __restrict__ fea)
{
    __shared__ float m4[128];
    int b = blockIdx.x, tid = threadIdx.x;           // 128 threads
    const float* yp = Y4 + ((size_t)b * 128 + tid) * 64;
    float s = 0.f;
    for (int p = 0; p < 64; ++p) s += yp[p];
    m4[tid] = s * (1.f / 64.f);
    __syncthreads();
    if (tid < 64) {
        float acc = bf[tid];
        const float* wr = wf + tid * 128;
        for (int c = 0; c < 128; ++c) acc = fmaf(wr[c], m4[c], acc);
        fea[b * 64 + tid] = acc;
    }
}

// ---------------------------------------------------------------------------
// Build per-sample folded params:
//   M1 = u0@v0 + I,  K1 = M1@cf_w (64x3), c1 = M1@cf_b + shift_first
//   M2 = u1@v1 + I,  K2 = M2@hr_w (64x64, stored f16 in WMMA A-fragment order),
//                    c2 = M2@hr_b + shift_HR
//   K3 = diag(1+sl)@cl_w (3x64),  c3 = (1+sl)*cl_b + shl
// One block (64 threads) per sample; thread i owns row i.
// ---------------------------------------------------------------------------
__global__ void dynk_kernel(const float* __restrict__ fea_g,
                            const float* w_sf0, const float* b_sf0,
                            const float* w_sf1, const float* b_sf1,
                            const float* w_hr0, const float* b_hr0,
                            const float* w_hr1, const float* b_hr1,
                            const float* w_sl,  const float* b_sl,
                            const float* w_shf, const float* b_shf,
                            const float* w_shr, const float* b_shr,
                            const float* w_shl, const float* b_shl,
                            const float* cf_w, const float* cf_b,
                            const float* hr_w, const float* hr_b,
                            const float* cl_w, const float* cl_b,
                            float* __restrict__ params, _Float16* __restrict__ k2frag)
{
    __shared__ float fs[64], u0[256], v0[256], u1[256], v1[256];
    __shared__ float shf[64], shr[64], sl3[4], shl3[4];
    __shared__ float m2s[64 * 64];
    int b = blockIdx.x, tid = threadIdx.x;           // 64 threads
    fs[tid] = fea_g[b * 64 + tid];
    __syncthreads();
    auto lin = [&](const float* w, const float* bb, int o) {
        float acc = bb[o];
        const float* wr = w + o * 64;
        for (int k = 0; k < 64; ++k) acc = fmaf(wr[k], fs[k], acc);
        return acc;
    };
    for (int o = tid; o < 256; o += 64) {
        u0[o] = lin(w_sf0, b_sf0, o);
        v0[o] = lin(w_sf1, b_sf1, o);
        u1[o] = lin(w_hr0, b_hr0, o);
        v1[o] = lin(w_hr1, b_hr1, o);
    }
    shf[tid] = lin(w_shf, b_shf, tid);
    shr[tid] = lin(w_shr, b_shr, tid);
    if (tid < 3) { sl3[tid] = lin(w_sl, b_sl, tid); shl3[tid] = lin(w_shl, b_shl, tid); }
    __syncthreads();

    float* pp = params + b * PSTRIDE;
    int i = tid;
    // --- row i of M1 -> K1 row + c1 ---
    {
        float a0 = u0[i*4+0], a1 = u0[i*4+1], a2 = u0[i*4+2], a3 = u0[i*4+3];
        float k10 = 0.f, k11 = 0.f, k12 = 0.f, c1 = shf[i];
        for (int j = 0; j < 64; ++j) {
            float m = a0*v0[j] + a1*v0[64+j] + a2*v0[128+j] + a3*v0[192+j] + (i == j ? 1.f : 0.f);
            k10 = fmaf(m, cf_w[j*3+0], k10);
            k11 = fmaf(m, cf_w[j*3+1], k11);
            k12 = fmaf(m, cf_w[j*3+2], k12);
            c1  = fmaf(m, cf_b[j], c1);
        }
        pp[i*3+0] = k10; pp[i*3+1] = k11; pp[i*3+2] = k12; pp[192 + i] = c1;
    }
    // --- row i of M2 -> m2s + c2 ---
    {
        float a0 = u1[i*4+0], a1 = u1[i*4+1], a2 = u1[i*4+2], a3 = u1[i*4+3];
        float c2 = shr[i];
        for (int j = 0; j < 64; ++j) {
            float m = a0*v1[j] + a1*v1[64+j] + a2*v1[128+j] + a3*v1[192+j] + (i == j ? 1.f : 0.f);
            m2s[i*64 + j] = m;
            c2 = fmaf(m, hr_b[j], c2);
        }
        pp[256 + i] = c2;
    }
    // --- K2 row i = M2row @ hr_w, scattered into WMMA 16-bit A-fragment order.
    // contiguous-k A layout used here (B matches): lane<16  -> row, K {0..15}(j0..15)
    //                     lane>=16 -> row, K {16..31}(j0..15)
    _Float16* kf = k2frag + (size_t)b * 4096;
    int mt = i >> 4, lrow = i & 15;
    for (int c = 0; c < 64; ++c) {
        float acc = 0.f;
        for (int j = 0; j < 64; ++j) acc = fmaf(m2s[i*64 + j], hr_w[j*64 + c], acc);
        int kap = c >> 5, kk = c & 31;
        int lane, jj;
        if (kk < 16)      { lane = lrow;      jj = kk;      }
        else              { lane = 16 + lrow; jj = kk - 16; }
        kf[(((mt * 2 + kap) * 32 + lane) << 4) + jj] = (_Float16)acc;
    }
    // --- K3 / c3 ---
    if (tid < 3) {
        float s = 1.f + sl3[tid];
        for (int m = 0; m < 64; ++m) pp[320 + tid*64 + m] = s * cl_w[tid*64 + m];
        pp[512 + tid] = s * cl_b[tid] + shl3[tid];
    }
}

// ---------------------------------------------------------------------------
// Fused main path: per block = 128 pixels of one sample, 8 wave32.
//   h1 = relu(K1 x + c1)  (VALU, 3->64)  -> LDS f16 [pix][ch]
//   h2 = relu(K2 h1 + c2) (WMMA f16 16x16x32, 4 Mtiles x 2 Ksteps / wave)
//   y  = K3 h2 + c3       (VALU, 64->3)
// ---------------------------------------------------------------------------
__global__ void __launch_bounds__(256)
main_fused_kernel(const float* __restrict__ content,
                  const float* __restrict__ params,
                  const _Float16* __restrict__ k2frag,
                  float* __restrict__ out)
{
    __shared__ __align__(32) _Float16 h1s[128 * 64];   // 16 KB
    __shared__ float h2s[128 * 64];                    // 32 KB
    int b = blockIdx.y;
    int pixbase = blockIdx.x * 128;
    int tid = threadIdx.x;
    const float* pp  = params + b * PSTRIDE;
    const float* x0p = content + ((size_t)b * 3 + 0) * HW + pixbase;
    const float* x1p = content + ((size_t)b * 3 + 1) * HW + pixbase;
    const float* x2p = content + ((size_t)b * 3 + 2) * HW + pixbase;

    // Stage A: h1
    for (int i = tid; i < 64 * 128; i += 256) {
        int c = i >> 7, p = i & 127;
        float h = pp[192 + c];
        h = fmaf(pp[c*3+0], x0p[p], h);
        h = fmaf(pp[c*3+1], x1p[p], h);
        h = fmaf(pp[c*3+2], x2p[p], h);
        h1s[p * 64 + c] = (_Float16)fmaxf(h, 0.f);
    }
    __syncthreads();

    // Stage B: h2 via WMMA (EXEC all-ones here: no divergence)
    int wv = tid >> 5, lane = tid & 31;
    int col = lane & 15, half = lane >> 4;
    const _Float16* bsrc = &h1s[(wv * 16 + col) * 64 + half * 16];
    v16h bf0 = *(const v16h*)(bsrc);        // K 0..31 slice for this lane-half
    v16h bf1 = *(const v16h*)(bsrc + 32);   // K 32..63 slice
    const _Float16* kf = k2frag + (size_t)b * 4096;
    for (int m = 0; m < 4; ++m) {
        v8f acc;
        #pragma unroll
        for (int r = 0; r < 8; ++r) acc[r] = pp[256 + m * 16 + r + half * 8];  // c2 bias
        v16h aA = *(const v16h*)(kf + (((m * 2 + 0) * 32 + lane) << 4));
        v16h aB = *(const v16h*)(kf + (((m * 2 + 1) * 32 + lane) << 4));
        acc = __builtin_amdgcn_wmma_f32_16x16x32_f16(false, aA, false, bf0,
                                                     (short)0, acc, false, false);
        acc = __builtin_amdgcn_wmma_f32_16x16x32_f16(false, aB, false, bf1,
                                                     (short)0, acc, false, false);
        float* dst = &h2s[(wv * 16 + col) * 64 + m * 16 + half * 8];
        #pragma unroll
        for (int r = 0; r < 8; ++r) dst[r] = fmaxf(acc[r], 0.f);
    }
    __syncthreads();

    // Stage C: y = K3 h2 + c3
    for (int i = tid; i < 3 * 128; i += 256) {
        int ch = i >> 7, p = i & 127;
        float acc = pp[512 + ch];
        const float* k3 = pp + 320 + ch * 64;
        const float* hp = &h2s[p * 64];
        for (int m = 0; m < 64; ++m) acc = fmaf(k3[m], hp[m], acc);
        out[((size_t)b * 3 + ch) * HW + pixbase + p] = acc;
    }
}

// ---------------------------------------------------------------------------
extern "C" void kernel_launch(void* const* d_in, const int* in_sizes, int n_in,
                              void* d_out, int out_size, void* d_ws, size_t ws_size,
                              hipStream_t stream)
{
    (void)in_sizes; (void)n_in; (void)out_size; (void)ws_size;
    const float* content = (const float*)d_in[0];
    const float* cond    = (const float*)d_in[1];
    const float* cw[5] = { (const float*)d_in[2], (const float*)d_in[4], (const float*)d_in[6],
                           (const float*)d_in[8], (const float*)d_in[10] };
    const float* cb[5] = { (const float*)d_in[3], (const float*)d_in[5], (const float*)d_in[7],
                           (const float*)d_in[9], (const float*)d_in[11] };
    const float* gg[4] = { (const float*)d_in[12], (const float*)d_in[14],
                           (const float*)d_in[16], (const float*)d_in[18] };
    const float* ee[4] = { (const float*)d_in[13], (const float*)d_in[15],
                           (const float*)d_in[17], (const float*)d_in[19] };
    const float* wf = (const float*)d_in[20];
    const float* bf = (const float*)d_in[21];
    const float* w_sf0 = (const float*)d_in[22]; const float* b_sf0 = (const float*)d_in[23];
    const float* w_sf1 = (const float*)d_in[24]; const float* b_sf1 = (const float*)d_in[25];
    const float* w_hr0 = (const float*)d_in[26]; const float* b_hr0 = (const float*)d_in[27];
    const float* w_hr1 = (const float*)d_in[28]; const float* b_hr1 = (const float*)d_in[29];
    const float* w_sl  = (const float*)d_in[30]; const float* b_sl  = (const float*)d_in[31];
    const float* w_shf = (const float*)d_in[32]; const float* b_shf = (const float*)d_in[33];
    const float* w_shr = (const float*)d_in[34]; const float* b_shr = (const float*)d_in[35];
    const float* w_shl = (const float*)d_in[36]; const float* b_shl = (const float*)d_in[37];
    const float* cf_w = (const float*)d_in[38]; const float* cf_b = (const float*)d_in[39];
    const float* hr_w = (const float*)d_in[40]; const float* hr_b = (const float*)d_in[41];
    const float* cl_w = (const float*)d_in[42]; const float* cl_b = (const float*)d_in[43];

    // Workspace layout (floats). Total ~7.39 M floats (~29.6 MB).
    float* ws   = (float*)d_ws;
    float* Ybig = ws;                       // 4,194,304  (max stage output: 16x16x128x128)
    float* Ysml = ws + 4194304;             // 2,097,152
    float* P    = ws + 6291456;             // 1,048,576  (max pool output: 16x16x64x64)
    float* SUMS = ws + 7340032;             // 2048
    float* SSQS = ws + 7342080;             // 2048
    float* AFFA = ws + 7344128;             // 2048
    float* AFFT = ws + 7346176;             // 2048
    float* FEA  = ws + 7348224;             // 1024
    float* PAR  = ws + 7349248;             // 16 * PSTRIDE = 8320
    _Float16* K2F = (_Float16*)(ws + 7357568);  // 16 * 4096 f16

    const int chs[6] = { 3, 16, 32, 64, 128, 128 };
    const int res[6] = { 256, 128, 64, 32, 16, 8 };
    float* Yb[2] = { Ybig, Ysml };
    const float* stage_in = cond;
    for (int i = 0; i < 5; ++i) {
        int Ci = chs[i], Co = chs[i + 1];
        int Hin = res[i], Ho = res[i + 1];
        int npix = Ho * Ho;
        int totalP = BATCH * Ci * npix;
        pool_affine_kernel<<<(totalP + 255) / 256, 256, 0, stream>>>(
            stage_in, P, Ci, Hin, Hin, Ho, Ho,
            (i == 0) ? nullptr : AFFA, (i == 0) ? nullptr : AFFT);
        float* Ycur = Yb[i & 1];
        bool stats = (i < 4);
        if (stats) {
            hipMemsetAsync(SUMS, 0, BATCH * Co * sizeof(float), stream);
            hipMemsetAsync(SSQS, 0, BATCH * Co * sizeof(float), stream);
        }
        conv_lrelu_stats_kernel<<<dim3((npix + 255) / 256, Co, BATCH), 256, 0, stream>>>(
            P, cw[i], cb[i], Ycur, Ci, Co, Ho, Ho, Hin, Hin,
            stats ? SUMS : nullptr, stats ? SSQS : nullptr);
        if (stats)
            inorm_prep_kernel<<<(BATCH * Co + 255) / 256, 256, 0, stream>>>(
                SUMS, SSQS, gg[i], ee[i], AFFA, AFFT, Co, npix);
        stage_in = Ycur;
    }
    // Stage 4 output landed in Ybig (i=4 -> Yb[0]).
    fea_kernel<<<BATCH, 128, 0, stream>>>(Ybig, wf, bf, FEA);
    dynk_kernel<<<BATCH, 64, 0, stream>>>(FEA,
        w_sf0, b_sf0, w_sf1, b_sf1, w_hr0, b_hr0, w_hr1, b_hr1,
        w_sl, b_sl, w_shf, b_shf, w_shr, b_shr, w_shl, b_shl,
        cf_w, cf_b, hr_w, hr_b, cl_w, cl_b, PAR, K2F);
    main_fused_kernel<<<dim3(HW / 128, BATCH), 256, 0, stream>>>(
        content, PAR, K2F, (float*)d_out);
}
